// ComplexSSMState_7086696039112
// MI455X (gfx1250) — matrix operand order repaired and checked by reading, not compiled, not run on hardware
//
#include <hip/hip_runtime.h>
#include <hip/hip_bf16.h>
#include <math.h>

typedef __attribute__((ext_vector_type(16))) _Float16 v16h;
typedef __attribute__((ext_vector_type(8)))  float    v8f;

#define B_ 4
#define S_ 2048
#define G_ 8
#define D_ 128
#define N_ 64
#define BSG (B_*S_*G_)            // 65536 tokens
#define MELEM ((size_t)BSG * N_)  // 4194304 elems per (B,S,G,N) array
#define CHUNK 256                 // == RENORM_PERIOD
#define NCHUNK (S_/CHUNK)         // 8
#define NSCAN (B_*G_*N_)          // 2048 independent recurrence lanes
#define NCB (NSCAN*NCHUNK)        // 16384 (lane, chunk) pairs
#define STRIDE ((size_t)G_*N_)    // 512 floats between successive timesteps
#define PFD 32                    // prefetch distance in timesteps (64KB/stream)

// ---------------------------------------------------------------- WMMA utils

__device__ __forceinline__ v8f wmma_f16(v16h a, v16h b, v8f c) {
  // D = A(16x32 f16) * B(32x16 f16) + C(16x16 f32)
  return __builtin_amdgcn_wmma_f32_16x16x32_f16(
      /*neg_a=*/false, a, /*neg_b=*/false, b,
      /*c_mod=*/(short)0, c, /*reuse_a=*/false, /*reuse_b=*/false);
}

// A operand (16x32, f16): row = lane&15 ; half hh = lane>>4.
// ISA layout: VGPR0..3 hold K = 8*hh + {0..7}, VGPR4..7 hold K = 16 + 8*hh + {0..7}.
__device__ __forceinline__ v16h load_A16(const _Float16* row, int hh, int kb) {
  v16h a;
#pragma unroll
  for (int j = 0; j < 8; ++j) {
    int kk = kb + ((j < 4) ? (8*hh + 2*j) : (16 + 8*hh + 2*(j - 4)));
    a[2*j]     = row[kk];
    a[2*j + 1] = row[kk + 1];
  }
  return a;
}

// B operand (32x16, f16): col = lane&15 ; half hh = lane>>4.
// Same K order as A: VGPR0..3 hold K = 8*hh + {0..7}, VGPR4..7 hold K = 16 + 8*hh + {0..7}.
// `col` pointer must be contiguous over K.
__device__ __forceinline__ v16h load_B16(const _Float16* col, int hh, int kb) {
  v16h b;
#pragma unroll
  for (int j = 0; j < 8; ++j) {
    int kk = kb + ((j < 4) ? (8*hh + 2*j) : (16 + 8*hh + 2*(j - 4)));
    b[2*j]     = col[kk];
    b[2*j + 1] = col[kk + 1];
  }
  return b;
}

// ------------------------------------------------- Kernel A: dt + A + Bx (WMMA)
// One wave = one 16-token tile. Block = 64 threads (2 waves) -> ~53.5KB LDS.

__global__ __launch_bounds__(64)
void k_proj_in(const float* __restrict__ xr,  const float* __restrict__ xi,
               const float* __restrict__ logA, const float* __restrict__ Aph,
               const float* __restrict__ Bwr,  const float* __restrict__ Bwi,
               const float* __restrict__ dtw,  const float* __restrict__ dtb,
               float* __restrict__ Ar, float* __restrict__ Ai,
               float* __restrict__ Hr, float* __restrict__ Hi)
{
  __shared__ _Float16 sBwr[N_*D_], sBwi[N_*D_];   // Bw in f16, (n, d) row-major
  __shared__ float    sNL[G_*N_], sPH[G_*N_];     // -softplus(logA), A_phase
  __shared__ _Float16 sXr[2][16*D_], sXi[2][16*D_]; // per-wave x tile, f16
  __shared__ float    sDT[2][32];                  // per-wave dt: [0..15]=mag [16..31]=phase

  const int tid = threadIdx.x;
  for (int i = tid; i < N_*D_; i += 64) {
    sBwr[i] = (_Float16)Bwr[i];
    sBwi[i] = (_Float16)Bwi[i];
  }
  for (int i = tid; i < G_*N_; i += 64) {
    sNL[i] = -log1pf(__expf(logA[i]));  // neg_log_A = -softplus(log_A_mag)
    sPH[i] = Aph[i];
  }
  __syncthreads();

  const int w = tid >> 5, lane = tid & 31;
  const int m16 = lane & 15, hh = lane >> 4;
  const int tile = blockIdx.x*2 + w;
  const int token0 = tile*16;

  // Stage 16x128 x tile (coalesced f32 -> f16 in LDS)
  {
    const float4* xr4 = (const float4*)(xr + (size_t)token0 * D_);
    const float4* xi4 = (const float4*)(xi + (size_t)token0 * D_);
    for (int i = lane; i < 16*D_/4; i += 32) {
      float4 a = xr4[i]; float4 b = xi4[i];
      int e = 4*i;
      sXr[w][e+0] = (_Float16)a.x; sXr[w][e+1] = (_Float16)a.y;
      sXr[w][e+2] = (_Float16)a.z; sXr[w][e+3] = (_Float16)a.w;
      sXi[w][e+0] = (_Float16)b.x; sXi[w][e+1] = (_Float16)b.y;
      sXi[w][e+2] = (_Float16)b.z; sXi[w][e+3] = (_Float16)b.w;
    }
  }

  // dt: lane (m, o=hh) -> 256-long fp32 dot (reads hit L0/L2, x just loaded)
  {
    const int token = token0 + m16;
    const float* xrrow = xr + (size_t)token * D_;
    const float* xirow = xi + (size_t)token * D_;
    const float* wrow  = dtw + hh * (2*D_);
    float acc = dtb[hh];
#pragma unroll 4
    for (int d = 0; d < D_; ++d)
      acc = fmaf(xrrow[d], wrow[d], fmaf(xirow[d], wrow[D_ + d], acc));
    sDT[w][hh*16 + m16] = fminf(fmaxf(__expf(acc), 1e-4f), 2.0f);
  }
  __syncthreads();

  const _Float16* arow_r = sXr[w] + m16 * D_;
  const _Float16* arow_i = sXi[w] + m16 * D_;

#pragma unroll
  for (int nt = 0; nt < 4; ++nt) {
    v8f accR = {};  // x_r.Bw_r - x_i.Bw_i
    v8f accI = {};  // x_r.Bw_i + x_i.Bw_r
    const int n = m16 + nt*16;
    const _Float16* bcol_r = sBwr + n * D_;  // contiguous over k=d
    const _Float16* bcol_i = sBwi + n * D_;
#pragma unroll
    for (int ks = 0; ks < 4; ++ks) {  // K = 128 = 4 * 32
      const int kb = ks * 32;
      v16h a_r  = load_A16(arow_r, hh, kb);
      v16h a_i  = load_A16(arow_i, hh, kb);
      v16h a_ni = -a_i;                       // f16 WMMA has no A-neg modifier
      v16h b_r  = load_B16(bcol_r, hh, kb);
      v16h b_i  = load_B16(bcol_i, hh, kb);
      accR = wmma_f16(a_r,  b_r, accR);
      accR = wmma_f16(a_ni, b_i, accR);
      accI = wmma_f16(a_r,  b_i, accI);
      accI = wmma_f16(a_i,  b_r, accI);
    }
    // Epilogue: scale by dt_mag, compute A_r/A_i, store all four arrays
#pragma unroll
    for (int v = 0; v < 8; ++v) {
      const int mm    = v + 8*hh;        // C/D layout: VGPR v, half hh -> row
      const int token = token0 + mm;
      const int g     = token & (G_ - 1);
      const float dtm = sDT[w][mm];
      const float dtp = sDT[w][16 + mm];
      const float nl  = sNL[g*N_ + n];
      const float ph  = sPH[g*N_ + n];
      const float Am  = __expf(dtm * nl);
      const float ang = dtp * ph;
      const size_t o  = (size_t)token * N_ + n;
      Ar[o] = Am * __cosf(ang);
      Ai[o] = Am * __sinf(ang);
      Hr[o] = accR[v] * dtm;   // Bx_r * dt_mag  (u-input of the scan)
      Hi[o] = accI[v] * dtm;   // Bx_i * dt_mag
    }
  }
}

// ---------------------- Kernel B1: per-chunk zero-state scan + prefix product
// 16384 threads, each walks 256 steps of one (b,g,n,chunk). Writes u in place.
// Streams are 2KB-strided; prefetch ~64KB ahead to keep HBM queues full.

__global__ __launch_bounds__(256)
void k_scan_local(const float* __restrict__ Ar, const float* __restrict__ Ai,
                  float* __restrict__ Hr, float* __restrict__ Hi,
                  float* __restrict__ Pr, float* __restrict__ Pi)
{
  const int tid = blockIdx.x*256 + threadIdx.x;
  const int n = tid & 63, g = (tid >> 6) & 7, b = (tid >> 9) & 3, c = tid >> 11;
  size_t idx = (((size_t)b*S_ + (size_t)c*CHUNK)*G_ + g)*N_ + n;
  float hr = 0.f, hi = 0.f, pr = 1.f, pi = 0.f;
  for (int t0 = 0; t0 < CHUNK; t0 += 8) {
    // speculative prefetch (global_prefetch_b8); overshoot stays inside ws
    __builtin_prefetch(Ar + idx + PFD*STRIDE, 0, 0);
    __builtin_prefetch(Ai + idx + PFD*STRIDE, 0, 0);
    __builtin_prefetch(Hr + idx + PFD*STRIDE, 0, 0);
    __builtin_prefetch(Hi + idx + PFD*STRIDE, 0, 0);
#pragma unroll
    for (int t = 0; t < 8; ++t, idx += STRIDE) {
      const float ar = Ar[idx], ai = Ai[idx];
      const float br = Hr[idx], bi = Hi[idx];
      const float nr = fmaf(ar, hr, fmaf(-ai, hi, br));
      const float ni = fmaf(ar, hi, fmaf( ai, hr, bi));
      hr = nr; hi = ni;
      Hr[idx] = hr; Hi[idx] = hi;            // u_t (zero-state local solution)
      const float qr = ar*pr - ai*pi;
      const float qi = ar*pi + ai*pr;
      pr = qr; pi = qi;                       // P_chunk = prod a_t
    }
  }
  Pr[tid] = pr; Pi[tid] = pi;
}

// ------------- Kernel B2: sequential carry over the 8 chunks (+renorm at ends)

__global__ __launch_bounds__(256)
void k_scan_carry(const float* __restrict__ Hr, const float* __restrict__ Hi,
                  const float* __restrict__ Pr, const float* __restrict__ Pi,
                  float* __restrict__ C0r, float* __restrict__ C0i)
{
  const int t2 = blockIdx.x*256 + threadIdx.x;   // 0..2047 -> (b,g,n)
  const int n = t2 & 63, g = (t2 >> 6) & 7, b = t2 >> 9;
  float hr = 0.f, hi = 0.f;
  for (int c = 0; c < NCHUNK; ++c) {
    const int ci = (c << 11) | t2;
    C0r[ci] = hr; C0i[ci] = hi;                  // chunk initial state
    const size_t idx =
        (((size_t)b*S_ + (size_t)c*CHUNK + (CHUNK-1))*G_ + g)*N_ + n;
    const float pr = Pr[ci], pi = Pi[ci];
    const float er = pr*hr - pi*hi + Hr[idx];    // h at chunk end
    const float ei = pr*hi + pi*hr + Hi[idx];
    const float norm  = sqrtf(er*er + ei*ei + 1e-8f);
    const float scale = fminf(norm, 100.0f) / norm;  // renorm at (t+1)%256==0
    hr = er * scale; hi = ei * scale;
  }
}

// -------- Kernel B3: fixup h_t = P_t * h0 + u_t (recompute P_t on the fly)

__global__ __launch_bounds__(256)
void k_scan_fix(const float* __restrict__ Ar, const float* __restrict__ Ai,
                const float* __restrict__ C0r, const float* __restrict__ C0i,
                float* __restrict__ Hr, float* __restrict__ Hi)
{
  const int tid = blockIdx.x*256 + threadIdx.x;
  const int n = tid & 63, g = (tid >> 6) & 7, b = (tid >> 9) & 3, c = tid >> 11;
  const float h0r = C0r[tid], h0i = C0i[tid];
  size_t idx = (((size_t)b*S_ + (size_t)c*CHUNK)*G_ + g)*N_ + n;
  float pr = 1.f, pi = 0.f;
  for (int t0 = 0; t0 < CHUNK; t0 += 8) {
    __builtin_prefetch(Ar + idx + PFD*STRIDE, 0, 0);
    __builtin_prefetch(Ai + idx + PFD*STRIDE, 0, 0);
    __builtin_prefetch(Hr + idx + PFD*STRIDE, 0, 0);
    __builtin_prefetch(Hi + idx + PFD*STRIDE, 0, 0);
#pragma unroll
    for (int t = 0; t < 8; ++t, idx += STRIDE) {
      const float ar = Ar[idx], ai = Ai[idx];
      const float npr = ar*pr - ai*pi;
      const float npi = ar*pi + ai*pr;
      pr = npr; pi = npi;
      float hr = fmaf(pr, h0r, fmaf(-pi, h0i, Hr[idx]));
      float hi = fmaf(pr, h0i, fmaf( pi, h0r, Hi[idx]));
      if (t0 + t == CHUNK - 1) {
        const float norm  = sqrtf(hr*hr + hi*hi + 1e-8f);
        const float scale = fminf(norm, 100.0f) / norm;
        hr *= scale; hi *= scale;
      }
      Hr[idx] = hr; Hi[idx] = hi;
    }
  }
}

// ------------------------------------- Kernel C: y = h @ Cw (complex, WMMA)
// One wave = one 16-token tile, all 128 D columns. Block = 128 (~48KB LDS).

__global__ __launch_bounds__(128)
void k_proj_out(const float* __restrict__ Hr, const float* __restrict__ Hi,
                const float* __restrict__ Cwr, const float* __restrict__ Cwi,
                float* __restrict__ out)
{
  __shared__ _Float16 sCr[D_*N_], sCi[D_*N_];       // Cw (d, n) row-major f16
  __shared__ _Float16 sHr[4][16*N_], sHi[4][16*N_]; // per-wave h tile f16

  const int tid = threadIdx.x;
  for (int i = tid; i < D_*N_; i += 128) {
    sCr[i] = (_Float16)Cwr[i];
    sCi[i] = (_Float16)Cwi[i];
  }

  const int w = tid >> 5, lane = tid & 31;
  const int m16 = lane & 15, hh = lane >> 4;
  const int tile = blockIdx.x*4 + w;
  const int token0 = tile*16;

  {
    const float4* hr4 = (const float4*)(Hr + (size_t)token0 * N_);
    const float4* hi4 = (const float4*)(Hi + (size_t)token0 * N_);
    for (int i = lane; i < 16*N_/4; i += 32) {
      float4 a = hr4[i]; float4 b = hi4[i];
      int e = 4*i;
      sHr[w][e+0] = (_Float16)a.x; sHr[w][e+1] = (_Float16)a.y;
      sHr[w][e+2] = (_Float16)a.z; sHr[w][e+3] = (_Float16)a.w;
      sHi[w][e+0] = (_Float16)b.x; sHi[w][e+1] = (_Float16)b.y;
      sHi[w][e+2] = (_Float16)b.z; sHi[w][e+3] = (_Float16)b.w;
    }
  }
  __syncthreads();

  const size_t OUTH = (size_t)BSG * D_;   // offset of y_i half of output
  const _Float16* arow_r = sHr[w] + m16 * N_;
  const _Float16* arow_i = sHi[w] + m16 * N_;

#pragma unroll
  for (int dtile = 0; dtile < 8; ++dtile) {     // D = 128 = 8 * 16 columns
    v8f accR = {};  // h_r.Cw_r - h_i.Cw_i
    v8f accI = {};  // h_r.Cw_i + h_i.Cw_r
    const int dcol = m16 + dtile*16;
    const _Float16* ccol_r = sCr + dcol * N_;   // contiguous over k=n
    const _Float16* ccol_i = sCi + dcol * N_;
#pragma unroll
    for (int ks = 0; ks < 2; ++ks) {            // K = 64 = 2 * 32
      const int kb = ks * 32;
      v16h h_r  = load_A16(arow_r, hh, kb);
      v16h h_i  = load_A16(arow_i, hh, kb);
      v16h h_ni = -h_i;
      v16h c_r  = load_B16(ccol_r, hh, kb);
      v16h c_i  = load_B16(ccol_i, hh, kb);
      accR = wmma_f16(h_r,  c_r, accR);
      accR = wmma_f16(h_ni, c_i, accR);
      accI = wmma_f16(h_r,  c_i, accI);
      accI = wmma_f16(h_i,  c_r, accI);
    }
#pragma unroll
    for (int v = 0; v < 8; ++v) {
      const int token = token0 + v + 8*hh;
      const size_t o = (size_t)token * D_ + dcol;
      out[o]        = accR[v];   // y_r
      out[OUTH + o] = accI[v];   // y_i
    }
  }
}

// ------------------------------------------------------------------- launch

extern "C" void kernel_launch(void* const* d_in, const int* in_sizes, int n_in,
                              void* d_out, int out_size, void* d_ws, size_t ws_size,
                              hipStream_t stream) {
  (void)in_sizes; (void)n_in; (void)out_size; (void)ws_size;

  const float* xr   = (const float*)d_in[0];
  const float* xi   = (const float*)d_in[1];
  const float* logA = (const float*)d_in[2];
  const float* Aph  = (const float*)d_in[3];
  const float* Bwr  = (const float*)d_in[4];
  const float* Bwi  = (const float*)d_in[5];
  const float* Cwr  = (const float*)d_in[6];
  const float* Cwi  = (const float*)d_in[7];
  const float* dtw  = (const float*)d_in[8];
  const float* dtb  = (const float*)d_in[9];
  float* out = (float*)d_out;

  // Workspace layout (floats): 4 * MELEM + 4 * NCB  (~67.4 MB)
  float* ws  = (float*)d_ws;
  float* Ar  = ws;
  float* Ai  = Ar + MELEM;
  float* Hr  = Ai + MELEM;   // Bx -> u -> h  (reused in place)
  float* Hi  = Hr + MELEM;
  float* Pr  = Hi + MELEM;
  float* Pi  = Pr + NCB;
  float* C0r = Pi + NCB;
  float* C0i = C0r + NCB;

  k_proj_in  <<<BSG/16/2, 64,  0, stream>>>(xr, xi, logA, Aph, Bwr, Bwi,
                                            dtw, dtb, Ar, Ai, Hr, Hi);
  k_scan_local<<<NCB/256,  256, 0, stream>>>(Ar, Ai, Hr, Hi, Pr, Pi);
  k_scan_carry<<<NSCAN/256,256, 0, stream>>>(Hr, Hi, Pr, Pi, C0r, C0i);
  k_scan_fix <<<NCB/256,   256, 0, stream>>>(Ar, Ai, C0r, C0i, Hr, Hi);
  k_proj_out <<<BSG/16/4,  128, 0, stream>>>(Hr, Hi, Cwr, Cwi, out);
}
